// DecoderLayer_43877385896448
// MI455X (gfx1250) — matrix operand order repaired and checked by reading, not compiled, not run on hardware
//
#include <hip/hip_runtime.h>
#include <math.h>

// ---------------------------------------------------------------------------
// Types for WMMA (CDNA5 / gfx1250, wave32)
// ---------------------------------------------------------------------------
typedef __attribute__((ext_vector_type(16))) _Float16 v16h;
typedef __attribute__((ext_vector_type(8)))  _Float16 v8h;
typedef __attribute__((ext_vector_type(8)))  float    v8f;
typedef __attribute__((ext_vector_type(4)))  float    fvec4;
typedef unsigned int u32a __attribute__((may_alias));

// Pack two fp32 into one dword of two f16 via v_cvt_pk_rtz_f16_f32 (1 VALU op).
static __device__ inline unsigned pack2h(float a, float b)
{
    auto p = __builtin_amdgcn_cvt_pkrtz(a, b);   // __fp16 ext_vector_type(2)
    unsigned u;
    __builtin_memcpy(&u, &p, 4);
    return u;
}

// ---------------------------------------------------------------------------
// Tiled WMMA GEMM:  C[M,N] = epi(A[M,K] * W[K,N] + bias[N])
// A, W fp32 row-major in HBM; converted to f16 while staging into LDS.
// Block tile 128x128, BK=32, 256 threads = 8 waves, wave tile 64x32 (4x2 WMMA).
// Software-pipelined: next K-tile's global loads overlap current tile's WMMAs.
// EPI: 0 = bias, 1 = bias + relu
// ---------------------------------------------------------------------------
#define BM 128
#define BN 128
#define BK 32
#define LDSTR 40   // f16 elements per LDS row: 80B rows -> 16B-aligned frags

template<int EPI>
__global__ __launch_bounds__(256) void gemm_wmma_f16(
    const float* __restrict__ A, const float* __restrict__ W,
    const float* __restrict__ bias, float* __restrict__ C,
    int M, int N, int K)
{
    __shared__ __align__(16) _Float16 sA[BM * LDSTR];          // [row][k]
    __shared__ __align__(16) _Float16 sB[BN * LDSTR];          // [col][k] (transposed)

    const int tid  = threadIdx.x;
    const int lane = tid & 31;
    const int wave = tid >> 5;        // 0..7
    const int wm   = wave & 1;        // 2 waves over M -> 64 rows each
    const int wn   = wave >> 1;       // 4 waves over N -> 32 cols each
    const int m0   = blockIdx.y * BM;
    const int n0   = blockIdx.x * BN;

    v8f acc[4][2];
#pragma unroll
    for (int i = 0; i < 4; ++i)
#pragma unroll
        for (int j = 0; j < 2; ++j)
#pragma unroll
            for (int e = 0; e < 8; ++e) acc[i][j][e] = 0.0f;

    // Per-thread staging registers (one K-tile in flight).
    fvec4 aReg[4];            // A: 4 float4 (row i>>3, cols c4*4)
    fvec4 b0Reg[2], b1Reg[2]; // B: 2x two consecutive K-rows, same 4 cols

    auto stage_load = [&](int kt) {
#pragma unroll
        for (int r = 0; r < 4; ++r) {
            int i = tid + r * 256, row = i >> 3, c4 = i & 7;
            aReg[r] = *(const fvec4*)(A + (size_t)(m0 + row) * K + kt + c4 * 4);
        }
#pragma unroll
        for (int r = 0; r < 2; ++r) {
            int i = tid + r * 256, krow = (i >> 5) * 2, c4 = i & 31;
            const float* wp = W + (size_t)(kt + krow) * N + n0 + c4 * 4;
            b0Reg[r] = *(const fvec4*)(wp);
            b1Reg[r] = *(const fvec4*)(wp + N);
        }
    };

    auto stage_store = [&]() {
#pragma unroll
        for (int r = 0; r < 4; ++r) {
            int i = tid + r * 256, row = i >> 3, c4 = i & 7;
            int off = row * LDSTR + c4 * 4;
            *(u32a*)(sA + off)     = pack2h(aReg[r][0], aReg[r][1]);
            *(u32a*)(sA + off + 2) = pack2h(aReg[r][2], aReg[r][3]);
        }
#pragma unroll
        for (int r = 0; r < 2; ++r) {
            int i = tid + r * 256, krow = (i >> 5) * 2, c4 = i & 31;
#pragma unroll
            for (int e = 0; e < 4; ++e)
                *(u32a*)(sB + (c4 * 4 + e) * LDSTR + krow) =
                    pack2h(b0Reg[r][e], b1Reg[r][e]);
        }
    };

    stage_load(0);

    for (int kt = 0; kt < K; kt += BK) {
        __syncthreads();                 // previous tile's readers done
        stage_store();
        __syncthreads();                 // stores visible to all waves

        if (kt + BK < K) {
            stage_load(kt + BK);         // overlaps with WMMA compute below
            // speculative prefetch of the tile after next (global_prefetch_b8)
            __builtin_prefetch(A + (size_t)(m0 + (tid >> 3)) * K + kt + 2 * BK, 0, 1);
        }

        // ---- A fragments (ISA 16-bit A 16x32 layout):
        // lane&15 = row; lane>>4 selects K halves {0..7,16..23} / {8..15,24..31}
        const int arow = lane & 15;
        const int kh   = (lane >> 4) * 8;
        v16h afrag[4];
#pragma unroll
        for (int mt = 0; mt < 4; ++mt) {
            const _Float16* base = sA + (size_t)(wm * 64 + mt * 16 + arow) * LDSTR + kh;
            v8h lo = *(const v8h*)(base);         // K = kh .. kh+7
            v8h hi = *(const v8h*)(base + 16);    // K = kh+16 .. kh+23
            afrag[mt] = __builtin_shufflevector(lo, hi,
                0,1,2,3,4,5,6,7,8,9,10,11,12,13,14,15);
        }
        // ---- both B fragments up-front so all ds_loads issue before the WMMAs
        v16h bfrag[2];
#pragma unroll
        for (int nt = 0; nt < 2; ++nt) {
            const int col  = wn * 32 + nt * 16 + (lane & 15);
            const int kh16 = (lane >> 4) * 8;
            const _Float16* base = sB + (size_t)col * LDSTR + kh16;
            v8h b0 = *(const v8h*)(base);
            v8h b1 = *(const v8h*)(base + 16);
            bfrag[nt] = __builtin_shufflevector(b0, b1,
                0,1,2,3,4,5,6,7,8,9,10,11,12,13,14,15);
        }
#pragma unroll
        for (int nt = 0; nt < 2; ++nt)
#pragma unroll
            for (int mt = 0; mt < 4; ++mt)
                acc[mt][nt] = __builtin_amdgcn_wmma_f32_16x16x32_f16(
                    false, afrag[mt], false, bfrag[nt], (short)0, acc[mt][nt],
                    false, false);
    }

    // ---- epilogue: C/D layout — VGPR v: M = v + (lane>>4)*8, N = lane&15
#pragma unroll
    for (int mt = 0; mt < 4; ++mt)
#pragma unroll
        for (int nt = 0; nt < 2; ++nt) {
            const int col   = n0 + wn * 32 + nt * 16 + (lane & 15);
            const float bv  = bias[col];
            const int rbase = m0 + wm * 64 + mt * 16 + (lane >> 4) * 8;
#pragma unroll
            for (int v = 0; v < 8; ++v) {
                float r = acc[mt][nt][v] + bv;
                if (EPI == 1) r = fmaxf(r, 0.0f);
                C[(size_t)(rbase + v) * N + col] = r;
            }
        }
}

// ---------------------------------------------------------------------------
// Fused q/k projection (N = H = 16 per proj) + sigmoid, transposed output:
// sigq[b,h,s] = sigmoid(qin[b,s,:] @ wq[:,h] + bq[h]); same for k.
// One block per (b,s) row. D must be 1024 (256 threads x float4).
// ---------------------------------------------------------------------------
__global__ __launch_bounds__(256) void qk_proj_kernel(
    const float* __restrict__ qin, const float* __restrict__ kin,
    const float* __restrict__ wq, const float* __restrict__ bq,
    const float* __restrict__ wk, const float* __restrict__ bk,
    float* __restrict__ sigq, float* __restrict__ sigk,
    int S, int D, int H)
{
    __shared__ float qrow[1024];
    __shared__ float krow[1024];
    __shared__ float red[32][9];

    const int tid = threadIdx.x;
    const int row = blockIdx.x;           // b*S + s
    const int b   = row / S;
    const int sidx = row - b * S;

    *(fvec4*)(qrow + tid * 4) = *(const fvec4*)(qin + (size_t)row * D + tid * 4);
    *(fvec4*)(krow + tid * 4) = *(const fvec4*)(kin + (size_t)row * D + tid * 4);
    __syncthreads();

    const int out  = tid >> 3;            // 0..31: 16 q then 16 k
    const int part = tid & 7;
    const float* src = (out < 16) ? qrow : krow;
    const float* wgt = (out < 16) ? wq : wk;
    const int h = out & 15;

    float acc = 0.0f;
    const int k0 = part * 128;
    for (int k = k0; k < k0 + 128; ++k) acc += src[k] * wgt[k * H + h];
    red[out][part] = acc;
    __syncthreads();

    if (tid < 32) {
        float t = 0.0f;
#pragma unroll
        for (int p = 0; p < 8; ++p) t += red[tid][p];
        const int hh = tid & 15;
        if (tid < 16) {
            t += bq[hh];
            sigq[((size_t)b * H + hh) * S + sidx] = 1.0f / (1.0f + __expf(-t));
        } else {
            t += bk[hh];
            sigk[((size_t)b * H + hh) * S + sidx] = 1.0f / (1.0f + __expf(-t));
        }
    }
}

// ---------------------------------------------------------------------------
// kv[b,h,d] = sum_s sigk[b,h,s] * v[b,s,h*64+d].  One block per (b,h).
// ---------------------------------------------------------------------------
__global__ __launch_bounds__(256) void kv_reduce_kernel(
    const float* __restrict__ sigk, const float* __restrict__ V,
    float* __restrict__ kv, int S, int D, int H)
{
    __shared__ float red[256];
    const int bh = blockIdx.x;
    const int b  = bh >> 4;
    const int h  = bh & 15;
    const int tid = threadIdx.x;
    const int d   = tid & 63;
    const int sc  = tid >> 6;            // 4 S-chunks
    const int chunk = S >> 2;

    const float* sk = sigk + (size_t)bh * S;
    const float* vp = V + (size_t)b * S * D + h * 64 + d;

    float acc = 0.0f;
    for (int s = sc * chunk; s < (sc + 1) * chunk; ++s)
        acc += sk[s] * vp[(size_t)s * D];
    red[tid] = acc;
    __syncthreads();
    if (tid < 64)
        kv[(size_t)bh * 64 + d] = red[d] + red[64 + d] + red[128 + d] + red[192 + d];
}

// In-place cumulative sum over the HEAD axis (the reference's cumsum quirk).
__global__ void cumsum_heads_kernel(float* __restrict__ kv, int B, int H, int dv)
{
    const int t = threadIdx.x;
    if (t < B * dv) {
        const int b = t / dv, d = t - (t / dv) * dv;
        float run = 0.0f;
        for (int h = 0; h < H; ++h) {
            const size_t idx = ((size_t)(b * H + h)) * dv + d;
            run += kv[idx];
            kv[idx] = run;
        }
    }
}

// out[b,s,h*64+d] = sigq[b,h,s] * scum[b,h,d]   (fixed dims: S=4096, H=16, dv=64)
__global__ __launch_bounds__(256) void outer_kernel(
    const float* __restrict__ sigq, const float* __restrict__ scum,
    float* __restrict__ out)
{
    const unsigned n = blockIdx.x * 256u + threadIdx.x;   // < 2^24
    const unsigned d = n & 63u;
    const unsigned h = (n >> 6) & 15u;
    const unsigned s = (n >> 10) & 4095u;
    const unsigned b = n >> 22;
    out[n] = sigq[((size_t)(b * 16u + h)) * 4096u + s] *
             scum[((size_t)(b * 16u + h)) * 64u + d];
}

// out = LN(a + res) * g + beta.   One block per row, D = 1024.
__global__ __launch_bounds__(256) void add_ln_kernel(
    const float* __restrict__ a, const float* __restrict__ res,
    const float* __restrict__ g, const float* __restrict__ beta,
    float* __restrict__ out, int D)
{
    __shared__ float red[256];
    const int tid = threadIdx.x;
    const size_t base = (size_t)blockIdx.x * D;

    fvec4 av = *(const fvec4*)(a + base + tid * 4);
    fvec4 rv = *(const fvec4*)(res + base + tid * 4);
    float x[4];
#pragma unroll
    for (int e = 0; e < 4; ++e) x[e] = av[e] + rv[e];

    float s = x[0] + x[1] + x[2] + x[3];
    red[tid] = s; __syncthreads();
    for (int off = 128; off > 0; off >>= 1) {
        if (tid < off) red[tid] += red[tid + off];
        __syncthreads();
    }
    const float mu = red[0] / (float)D;
    __syncthreads();

    float sq = 0.0f;
#pragma unroll
    for (int e = 0; e < 4; ++e) { float t = x[e] - mu; sq += t * t; }
    red[tid] = sq; __syncthreads();
    for (int off = 128; off > 0; off >>= 1) {
        if (tid < off) red[tid] += red[tid + off];
        __syncthreads();
    }
    const float rstd = rsqrtf(red[0] / (float)D + 1e-6f);

    fvec4 gv = *(const fvec4*)(g + tid * 4);
    fvec4 bv = *(const fvec4*)(beta + tid * 4);
    fvec4 ov;
#pragma unroll
    for (int e = 0; e < 4; ++e) ov[e] = gv[e] * (x[e] - mu) * rstd + bv[e];
    *(fvec4*)(out + base + tid * 4) = ov;
}

__global__ void fill_zero_kernel(float* __restrict__ p, size_t n)
{
    size_t i = (size_t)blockIdx.x * 256u + threadIdx.x;
    if (i < n) p[i] = 0.0f;
}

// ---------------------------------------------------------------------------
// Host-side orchestration
// ---------------------------------------------------------------------------
static void run_gemm(const float* A, const float* W, const float* bias, float* C,
                     int M, int N, int K, int relu, hipStream_t stream)
{
    dim3 grid(N / BN, M / BM);
    if (relu)
        gemm_wmma_f16<1><<<grid, 256, 0, stream>>>(A, W, bias, C, M, N, K);
    else
        gemm_wmma_f16<0><<<grid, 256, 0, stream>>>(A, W, bias, C, M, N, K);
}

extern "C" void kernel_launch(void* const* d_in, const int* in_sizes, int n_in,
                              void* d_out, int out_size, void* d_ws, size_t ws_size,
                              hipStream_t stream)
{
    (void)in_sizes; (void)n_in; (void)ws_size;
    constexpr int B = 4, S = 4096, D = 1024, H = 16, DV = 64, F = 4096;
    constexpr int M = B * S;                       // 16384
    constexpr size_t BSD = (size_t)M * D;          // 16,777,216

    const float* x   = (const float*)d_in[0];
    const float* enc = (const float*)d_in[1];
    const float* wq1 = (const float*)d_in[2];  const float* bq1 = (const float*)d_in[3];
    const float* wk1 = (const float*)d_in[4];  const float* bk1 = (const float*)d_in[5];
    const float* wv1 = (const float*)d_in[6];  const float* bv1 = (const float*)d_in[7];
    const float* wo1 = (const float*)d_in[8];  const float* bo1 = (const float*)d_in[9];
    const float* wq2 = (const float*)d_in[10]; const float* bq2 = (const float*)d_in[11];
    const float* wk2 = (const float*)d_in[12]; const float* bk2 = (const float*)d_in[13];
    const float* wv2 = (const float*)d_in[14]; const float* bv2 = (const float*)d_in[15];
    const float* wo2 = (const float*)d_in[16]; const float* bo2 = (const float*)d_in[17];
    const float* wf1 = (const float*)d_in[18]; const float* bf1 = (const float*)d_in[19];
    const float* wf2 = (const float*)d_in[20]; const float* bf2 = (const float*)d_in[21];
    const float* g1  = (const float*)d_in[22]; const float* be1 = (const float*)d_in[23];
    const float* g2  = (const float*)d_in[24]; const float* be2 = (const float*)d_in[25];
    const float* g3  = (const float*)d_in[26]; const float* be3 = (const float*)d_in[27];

    // workspace layout (floats)
    float* ws = (float*)d_ws;
    size_t o = 0;
    float* sigq = ws + o; o += (size_t)B * H * S;   // 262,144
    float* sigk = ws + o; o += (size_t)B * H * S;
    float* vbuf = ws + o; o += BSD;
    float* scum = ws + o; o += (size_t)B * H * DV;  // 4,096 (kv, cumsum'd in place)
    float* pbuf = ws + o; o += BSD;                 // outer-product / GEMM input
    float* abuf = ws + o; o += BSD;                 // attn/ffn GEMM output
    float* out1 = ws + o; o += BSD;
    float* out2 = ws + o; o += BSD;
    float* ffnm = ws + o; o += (size_t)M * F;       // 67,108,864

    float* out3 = (float*)d_out;

    const dim3 rowsGrid(M);
    const dim3 elemGrid((unsigned)(BSD / 256));

    // ---------------- attention 1 (self): q=k=v from x ----------------
    qk_proj_kernel<<<rowsGrid, 256, 0, stream>>>(x, x, wq1, bq1, wk1, bk1,
                                                 sigq, sigk, S, D, H);
    run_gemm(x, wv1, bv1, vbuf, M, D, D, 0, stream);
    kv_reduce_kernel<<<dim3(B * H), 256, 0, stream>>>(sigk, vbuf, scum, S, D, H);
    cumsum_heads_kernel<<<1, 256, 0, stream>>>(scum, B, H, DV);
    outer_kernel<<<elemGrid, 256, 0, stream>>>(sigq, scum, pbuf);
    run_gemm(pbuf, wo1, bo1, abuf, M, D, D, 0, stream);
    add_ln_kernel<<<rowsGrid, 256, 0, stream>>>(abuf, x, g1, be1, out1, D);

    // ---------------- attention 2 (cross): q from out1, k/v from enc ----------------
    qk_proj_kernel<<<rowsGrid, 256, 0, stream>>>(out1, enc, wq2, bq2, wk2, bk2,
                                                 sigq, sigk, S, D, H);
    run_gemm(enc, wv2, bv2, vbuf, M, D, D, 0, stream);
    kv_reduce_kernel<<<dim3(B * H), 256, 0, stream>>>(sigk, vbuf, scum, S, D, H);
    cumsum_heads_kernel<<<1, 256, 0, stream>>>(scum, B, H, DV);
    outer_kernel<<<elemGrid, 256, 0, stream>>>(sigq, scum, pbuf);
    run_gemm(pbuf, wo2, bo2, abuf, M, D, D, 0, stream);
    add_ln_kernel<<<rowsGrid, 256, 0, stream>>>(abuf, out1, g2, be2, out2, D);

    // ---------------- FFN ----------------
    run_gemm(out2, wf1, bf1, ffnm, M, F, D, 1, stream);   // relu
    run_gemm(ffnm, wf2, bf2, abuf, M, D, F, 0, stream);
    add_ln_kernel<<<rowsGrid, 256, 0, stream>>>(abuf, out2, g3, be3, out3, D);

    // outputs 2 and 3 of the tuple are zeros
    size_t tail = (size_t)out_size - BSD;
    fill_zero_kernel<<<dim3((unsigned)((tail + 255) / 256)), 256, 0, stream>>>(
        out3 + BSD, tail);
}
